// IDGNN_19207093748620
// MI455X (gfx1250) — matrix operand order repaired and checked by reading, not compiled, not run on hardware
//
#include <hip/hip_runtime.h>

// ID-GNN, N=256 nodes, D=128, L=2 layers, E=2048 edges.
// Algebraic restructuring (see analysis): layer 0 collapses to per-node MLPs,
// layer 1 is one fused per-target WMMA kernel over the closed-form H1 tile.

#define N_NODES 256
#define D_FEAT  128
#define N_EDGES 2048

typedef __attribute__((ext_vector_type(16))) _Float16 v16h;
typedef __attribute__((ext_vector_type(8)))  float    v8f;

// ---------------------------------------------------------------------------
// WMMA fragment loaders (contiguous k order, same for A and B, wave32).
// A (16x32 f16): lane<16 holds row M=lane, K = k0+{0..15};
//                lane>=16 holds row M=lane-16, K = k0+{16..31}.
// ---------------------------------------------------------------------------
__device__ __forceinline__ v16h afrag(const _Float16* S, int row0, int k0, int lane) {
  const int h = lane >> 4, m = lane & 15;
  const _Float16* p = S + (row0 + m) * D_FEAT + k0 + 16 * h;
  v16h a;
#pragma unroll
  for (int i = 0; i < 16; ++i) a[i] = p[i];
  return a;
}

// B (32x16 f16): lane<16 holds col N=lane with K=k0..k0+15; lane>=16 holds
// col N=lane-16 with K=k0+16..k0+31. Weights are stored pre-transposed
// (Wt[n][k] = W[k][n]) so each lane reads 16 contiguous halfs (32B).
__device__ __forceinline__ v16h bfrag(const _Float16* Wt, int k0, int n0, int lane) {
  const int h = lane >> 4, n = lane & 15;
  const _Float16* p = Wt + (n0 + n) * D_FEAT + k0 + 16 * h;
  v16h b;
#pragma unroll
  for (int i = 0; i < 16; ++i) b[i] = p[i];
  return b;
}

// ---------------------------------------------------------------------------
// Prep: fp16 weights, transposed, with w1's two K-halves pre-summed
// (z = cat(H,H)  =>  z @ w1 == H @ (w1[:D] + w1[D:])).
// wf layout: mats 0..3 = w1s_t[l][p] (n,k), mats 4..7 = w2_t[l][p] (n,k).
// ---------------------------------------------------------------------------
__global__ void prep_weights_kernel(const float* __restrict__ w1,
                                    const float* __restrict__ w2,
                                    _Float16* __restrict__ wf) {
  const int id = blockIdx.x * blockDim.x + threadIdx.x;   // 8*16384
  const int mat = id >> 14;
  const int rem = id & 16383;
  const int n = rem >> 7;   // output column
  const int k = rem & 127;  // input row
  float v;
  if (mat < 4) {
    const float* base = w1 + mat * (2 * D_FEAT * D_FEAT);       // (2D, D)
    v = base[k * D_FEAT + n] + base[(D_FEAT + k) * D_FEAT + n];
  } else {
    const float* base = w2 + (mat - 4) * (D_FEAT * D_FEAT);     // (D, D)
    v = base[k * D_FEAT + n];
  }
  wf[id] = (_Float16)v;
}

// A := 0, M := I
__global__ void init_adj_mask_kernel(float* __restrict__ A, float* __restrict__ M0) {
  const int id = blockIdx.x * blockDim.x + threadIdx.x;   // 65536
  A[id] = 0.f;
  M0[id] = ((id >> 8) == (id & 255)) ? 1.f : 0.f;
}

// A[r,c] = A[c,r] = 1  (scatter + symmetrize in one pass; int64 edge indices)
__global__ void scatter_edges_kernel(const long long* __restrict__ ei,
                                     float* __restrict__ A) {
  const int e = blockIdx.x * blockDim.x + threadIdx.x;
  if (e < N_EDGES) {
    const int r = (int)ei[e];
    const int c = (int)ei[N_EDGES + e];
    A[r * N_NODES + c] = 1.f;
    A[c * N_NODES + r] = 1.f;
  }
}

// Mout = Min | ((Min @ A) > 0)   -- run twice for the 2-hop mask
__global__ void mask_hop_kernel(const float* __restrict__ Min,
                                const float* __restrict__ A,
                                float* __restrict__ Mout) {
  const int id = blockIdx.x * blockDim.x + threadIdx.x;   // 65536
  const int t = id >> 8, j = id & 255;
  const float* mr = Min + t * N_NODES;
  float s = 0.f;
  for (int n = 0; n < N_NODES; ++n) s += mr[n] * A[n * N_NODES + j];
  Mout[id] = (mr[j] != 0.f || s > 0.f) ? 1.f : 0.f;
}

// Layer-0 per-node MLPs: g_p[n] = relu(x[n] @ w1s[0,p] + b1[0,p]) @ w2[0,p] + b2[0,p]
__global__ void mlp_g_kernel(const float* __restrict__ x,
                             const _Float16* __restrict__ wf,
                             const float* __restrict__ b1,
                             const float* __restrict__ b2,
                             float* __restrict__ g0, float* __restrict__ g1) {
  const int n = blockIdx.x, p = blockIdx.y, tid = threadIdx.x;   // 128 threads
  __shared__ float xr[D_FEAT], hid[D_FEAT];
  xr[tid] = x[n * D_FEAT + tid];
  __syncthreads();
  const _Float16* W1 = wf + p * 16384;            // w1s_t[0][p]
  const _Float16* W2 = wf + (4 + p) * 16384;      // w2_t[0][p]
  float s = 0.f;
  const _Float16* wr = W1 + tid * D_FEAT;
  for (int k = 0; k < D_FEAT; ++k) s += xr[k] * (float)wr[k];
  s += b1[p * D_FEAT + tid];
  hid[tid] = s > 0.f ? s : 0.f;
  __syncthreads();
  s = 0.f;
  wr = W2 + tid * D_FEAT;
  for (int k = 0; k < D_FEAT; ++k) s += hid[k] * (float)wr[k];
  s += b2[p * D_FEAT + tid];
  (p == 0 ? g0 : g1)[n * D_FEAT + tid] = s;
}

// U = x + A @ g0 ; dh0 = g1 - g0
__global__ void aggregate0_kernel(const float* __restrict__ x,
                                  const float* __restrict__ A,
                                  const float* __restrict__ g0,
                                  const float* __restrict__ g1,
                                  float* __restrict__ U, float* __restrict__ dh0) {
  const int id = blockIdx.x * blockDim.x + threadIdx.x;   // 32768
  const int j = id >> 7, d = id & 127;
  const float* ar = A + j * N_NODES;
  float s = 0.f;
  for (int n = 0; n < N_NODES; ++n) s += ar[n] * g0[n * D_FEAT + d];
  U[id] = x[id] + s;
  dh0[id] = g1[id] - g0[id];
}

// ---------------------------------------------------------------------------
// Fused layer-1 + output kernel, one workgroup (8 wave32) per target t.
//   H1[t,n] = M[t,n] ? (U[n] + A[t,n]*dh0[t]) : 2*x[n]        (A symmetric)
//   out[t]  = H1[t,t] + sum_n A[t,n]*MLP10(H1[t,n])
//           + A[t,t]*(MLP11(H1[t,t]) - MLP10(H1[t,t]))
// Both GEMMs are 256x128 @ 128x128 done with v_wmma_f32_16x16x32_f16.
// ---------------------------------------------------------------------------
__launch_bounds__(256)
__global__ void idgnn_layer1_wmma_kernel(
    const float* __restrict__ x, const float* __restrict__ A,
    const float* __restrict__ M, const float* __restrict__ U,
    const float* __restrict__ dh0, const _Float16* __restrict__ wf,
    const float* __restrict__ b1, const float* __restrict__ b2,
    float* __restrict__ out) {
  const int t = blockIdx.x;
  const int tid = threadIdx.x;
  const int wave = tid >> 5, lane = tid & 31;

  __shared__ _Float16 Hs[N_NODES * D_FEAT];   // 64 KB: H1[t] tile (fp16)
  __shared__ _Float16 Ts[N_NODES * D_FEAT];   // 64 KB: post-ReLU hidden (fp16)
  __shared__ float sArow[N_NODES];
  __shared__ float sDh[D_FEAT], sAcc[D_FEAT], sHtt[D_FEAT];
  __shared__ float sHid[D_FEAT], sRow0[D_FEAT], sRow1[D_FEAT];

  if (tid < N_NODES) sArow[tid] = A[t * N_NODES + tid];
  if (tid < D_FEAT) { sDh[tid] = dh0[t * D_FEAT + tid]; sAcc[tid] = 0.f; }
  __syncthreads();

  // Build H1[t] (closed form from layer 0) into LDS as fp16.
  for (int idx = tid; idx < N_NODES * D_FEAT; idx += 256) {
    const int n = idx >> 7, k = idx & 127;
    const float m = M[t * N_NODES + n];
    const float v = (m != 0.f) ? (U[idx] + sArow[n] * sDh[k]) : (2.f * x[idx]);
    Hs[idx] = (_Float16)v;
  }
  __syncthreads();

  // GEMM1: Ts = relu(Hs @ W1 + b1[1,0])   (fp16 in, f32 accum, fp16 out)
  {
    const _Float16* W1a = wf + 2 * 16384;   // w1s_t[1][0]
    const float* b1a = b1 + 2 * D_FEAT;
#pragma unroll
    for (int mt = 0; mt < 2; ++mt) {
      const int row0 = wave * 32 + mt * 16;
      v16h a0 = afrag(Hs, row0, 0, lane);
      v16h a1 = afrag(Hs, row0, 32, lane);
      v16h a2 = afrag(Hs, row0, 64, lane);
      v16h a3 = afrag(Hs, row0, 96, lane);
      for (int nt = 0; nt < 8; ++nt) {
        const int col0 = nt << 4;
        v8f c = {};
        c = __builtin_amdgcn_wmma_f32_16x16x32_f16(false, a0, false, bfrag(W1a,  0, col0, lane), (short)0, c, false, false);
        c = __builtin_amdgcn_wmma_f32_16x16x32_f16(false, a1, false, bfrag(W1a, 32, col0, lane), (short)0, c, false, false);
        c = __builtin_amdgcn_wmma_f32_16x16x32_f16(false, a2, false, bfrag(W1a, 64, col0, lane), (short)0, c, false, false);
        c = __builtin_amdgcn_wmma_f32_16x16x32_f16(false, a3, false, bfrag(W1a, 96, col0, lane), (short)0, c, false, false);
        const int col = col0 + (lane & 15);
        const float bias = b1a[col];
        const int rbase = row0 + ((lane >> 4) << 3);
#pragma unroll
        for (int r = 0; r < 8; ++r) {
          const float v = c[r] + bias;
          Ts[(rbase + r) * D_FEAT + col] = (_Float16)(v > 0.f ? v : 0.f);
        }
      }
    }
  }
  __syncthreads();

  // GEMM2: h0 = Ts @ W2 + b2[1,0]; epilogue folds the A-weighted row sum.
  {
    const _Float16* W2a = wf + 6 * 16384;   // w2_t[1][0]
    const float* b2a = b2 + 2 * D_FEAT;
#pragma unroll
    for (int mt = 0; mt < 2; ++mt) {
      const int row0 = wave * 32 + mt * 16;
      v16h a0 = afrag(Ts, row0, 0, lane);
      v16h a1 = afrag(Ts, row0, 32, lane);
      v16h a2 = afrag(Ts, row0, 64, lane);
      v16h a3 = afrag(Ts, row0, 96, lane);
      for (int nt = 0; nt < 8; ++nt) {
        const int col0 = nt << 4;
        v8f c = {};
        c = __builtin_amdgcn_wmma_f32_16x16x32_f16(false, a0, false, bfrag(W2a,  0, col0, lane), (short)0, c, false, false);
        c = __builtin_amdgcn_wmma_f32_16x16x32_f16(false, a1, false, bfrag(W2a, 32, col0, lane), (short)0, c, false, false);
        c = __builtin_amdgcn_wmma_f32_16x16x32_f16(false, a2, false, bfrag(W2a, 64, col0, lane), (short)0, c, false, false);
        c = __builtin_amdgcn_wmma_f32_16x16x32_f16(false, a3, false, bfrag(W2a, 96, col0, lane), (short)0, c, false, false);
        const int col = col0 + (lane & 15);
        const float bias = b2a[col];
        const int rbase = row0 + ((lane >> 4) << 3);
#pragma unroll
        for (int r = 0; r < 8; ++r) {
          const int row = rbase + r;
          const float aw = sArow[row];
          if (aw != 0.f) atomicAdd(&sAcc[col], aw * (c[r] + bias));  // ds_add_f32
        }
      }
    }
  }
  __syncthreads();

  // Diagonal rows in f32 (tiny): H1[t,t], MLP10 and MLP11 of it.
  const float Att = sArow[t];
  if (tid < D_FEAT) sHtt[tid] = U[t * D_FEAT + tid] + Att * sDh[tid];
  __syncthreads();
  for (int p = 0; p < 2; ++p) {
    const _Float16* Wa = wf + (2 + p) * 16384;   // w1s_t[1][p]
    const _Float16* Wb = wf + (6 + p) * 16384;   // w2_t[1][p]
    const float* ba = b1 + (2 + p) * D_FEAT;
    const float* bb = b2 + (2 + p) * D_FEAT;
    if (tid < D_FEAT) {
      float s = 0.f;
      const _Float16* wr = Wa + tid * D_FEAT;
      for (int k = 0; k < D_FEAT; ++k) s += sHtt[k] * (float)wr[k];
      s += ba[tid];
      sHid[tid] = s > 0.f ? s : 0.f;
    }
    __syncthreads();
    if (tid < D_FEAT) {
      float s = 0.f;
      const _Float16* wr = Wb + tid * D_FEAT;
      for (int k = 0; k < D_FEAT; ++k) s += sHid[k] * (float)wr[k];
      (p == 0 ? sRow0 : sRow1)[tid] = s + bb[tid];
    }
    __syncthreads();
  }

  if (tid < D_FEAT)
    out[t * D_FEAT + tid] = sHtt[tid] + sAcc[tid] + Att * (sRow1[tid] - sRow0[tid]);
}

// ---------------------------------------------------------------------------
extern "C" void kernel_launch(void* const* d_in, const int* in_sizes, int n_in,
                              void* d_out, int out_size, void* d_ws, size_t ws_size,
                              hipStream_t stream) {
  const float* x  = (const float*)d_in[0];
  const float* w1 = (const float*)d_in[1];
  const float* b1 = (const float*)d_in[2];
  const float* w2 = (const float*)d_in[3];
  const float* b2 = (const float*)d_in[4];
  const long long* ei = (const long long*)d_in[5];   // int64 edge_index (2,E)

  char* ws = (char*)d_ws;
  _Float16* wf = (_Float16*)ws; ws += 8 * 16384 * sizeof(_Float16); // 256 KB
  float* A   = (float*)ws; ws += 65536 * sizeof(float);             // 256 KB
  float* M0  = (float*)ws; ws += 65536 * sizeof(float);
  float* M1  = (float*)ws; ws += 65536 * sizeof(float);
  float* g0  = (float*)ws; ws += 32768 * sizeof(float);
  float* g1  = (float*)ws; ws += 32768 * sizeof(float);
  float* U   = (float*)ws; ws += 32768 * sizeof(float);
  float* dh0 = (float*)ws; ws += 32768 * sizeof(float);
  (void)in_sizes; (void)n_in; (void)out_size; (void)ws_size;

  prep_weights_kernel<<<8 * 16384 / 256, 256, 0, stream>>>(w1, w2, wf);
  init_adj_mask_kernel<<<65536 / 256, 256, 0, stream>>>(A, M0);
  scatter_edges_kernel<<<(N_EDGES + 255) / 256, 256, 0, stream>>>(ei, A);
  mask_hop_kernel<<<65536 / 256, 256, 0, stream>>>(M0, A, M1);   // hop 1
  mask_hop_kernel<<<65536 / 256, 256, 0, stream>>>(M1, A, M0);   // hop 2
  mlp_g_kernel<<<dim3(N_NODES, 2), 128, 0, stream>>>(x, wf, b1, b2, g0, g1);
  aggregate0_kernel<<<32768 / 256, 256, 0, stream>>>(x, A, g0, g1, U, dh0);
  idgnn_layer1_wmma_kernel<<<N_NODES, 256, 0, stream>>>(x, A, M0, U, dh0, wf,
                                                        b1, b2, (float*)d_out);
}
